// LocalPoolPointNet_38147899523268
// MI455X (gfx1250) — compile-verified, matrix operand order checked by reading, not run on hardware
//
#include <hip/hip_runtime.h>

// ---------------------------------------------------------------------------
// LocalPoolPointNet for MI455X (gfx1250, wave32, WMMA + async global->LDS).
//
// Pipeline (per kernel_launch):
//   stage0: pos-MLP + ResNet block0 (WMMA)           -> net[B*N][32] (ws)
//   4x: fill seg(-inf) -> scatter-max -> stage_i (async gather+concat+WMMA)
//   fill d_out(0) -> final: net@W_c+b_c (WMMA), clamp0, atomic-umax scatter
//
// seg buffer [B,R3,32] f32 = 134.2 MB reuses d_out (same size); it fits in
// the 192 MB L2, so all scatter/gather atomic traffic is L2-resident.
// Weights and activation rows are staged into LDS with the CDNA5 async
// global->LDS DMA path (ASYNCcnt) when available; WMMA fragments are built
// from LDS (ds loads) instead of per-wave global loads.
// ---------------------------------------------------------------------------

#if __has_builtin(__builtin_amdgcn_global_load_async_to_lds_b128) && \
    __has_builtin(__builtin_amdgcn_s_wait_asynccnt)
#define USE_ASYNC_LDS 1
#else
#define USE_ASYNC_LDS 0
#endif

typedef __attribute__((ext_vector_type(16))) _Float16 v16h;
typedef __attribute__((ext_vector_type(8)))  float    v8f;
typedef int gv4i __attribute__((vector_size(16)));   // builtin's b128 unit

static constexpr int B_  = 4;
static constexpr int N_  = 100000;
static constexpr int R_  = 64;
static constexpr int R3_ = R_ * R_ * R_;   // 262144
static constexpr int HID = 32;

// 16-byte global->LDS copy: CDNA5 async DMA (ASYNCcnt) if available.
// Address-space casts go through integers (the ROCm-library idiom) because
// clang rejects direct generic<->AS casts; on gfx1250 a generic LDS pointer's
// low 32 bits ARE the LDS offset.
__device__ __forceinline__ void async_copy16(const float* g, float* l) {
#if USE_ASYNC_LDS
  __builtin_amdgcn_global_load_async_to_lds_b128(
      (__attribute__((address_space(1))) gv4i*)(unsigned long long)g,
      (__attribute__((address_space(3))) gv4i*)(unsigned int)(unsigned long long)l,
      /*offset=*/0, /*cpol=*/0);
#else
  *(float4*)l = *(const float4*)g;
#endif
}

__device__ __forceinline__ void async_wait_all() {
#if USE_ASYNC_LDS
  __builtin_amdgcn_s_wait_asynccnt(0);
#endif
}

__device__ __forceinline__ v8f wmma_f16(v16h a, v16h b, v8f c) {
  // D = A(16x32 f16) * B(32x16 f16) + C(16x16 f32)
  return __builtin_amdgcn_wmma_f32_16x16x32_f16(
      /*neg_a=*/false, a, /*neg_b=*/false, b,
      /*c_mod=*/(short)0, c, /*reuse_a=*/false, /*reuse_b=*/false);
}

// A-fragment (16x32, f16) built from f32 LDS row-major x[16][stride], K-block kb.
// ISA layout: lanes 0-15 -> M=lane,kHalf=0 ; lanes 16-31 -> M=lane-16,kHalf=8
//   v0..3 : K = kHalf + 2v, +1 ; v4..7 : K = 16 + kHalf + 2(v-4), +1
__device__ __forceinline__ v16h load_A_lds(const float* x, int stride,
                                           int kb, bool relu) {
  const int lane  = threadIdx.x & 31;
  const int row   = lane & 15;
  const int khalf = (lane < 16) ? 0 : 8;
  v16h a;
#pragma unroll
  for (int v = 0; v < 8; ++v) {
    const int k = kb + ((v < 4) ? (khalf + 2 * v) : (16 + khalf + 2 * (v - 4)));
    float e0 = x[row * stride + k];
    float e1 = x[row * stride + k + 1];
    if (relu) { e0 = e0 > 0.f ? e0 : 0.f; e1 = e1 > 0.f ? e1 : 0.f; }
    a[2 * v]     = (_Float16)e0;
    a[2 * v + 1] = (_Float16)e1;
  }
  return a;
}

// B-fragment (32x16, f16) from f32 LDS row-major W[K][32], K-block kb, col n0.
// Mirror of A layout with N in place of M (lanes carry columns).
__device__ __forceinline__ v16h load_B_lds(const float* W, int kb, int n0) {
  const int lane  = threadIdx.x & 31;
  const int col   = n0 + (lane & 15);
  const int khalf = (lane < 16) ? 0 : 8;
  v16h b;
#pragma unroll
  for (int v = 0; v < 8; ++v) {
    const int k = kb + ((v < 4) ? (khalf + 2 * v) : (16 + khalf + 2 * (v - 4)));
    b[2 * v]     = (_Float16)W[k * 32 + col];
    b[2 * v + 1] = (_Float16)W[(k + 1) * 32 + col];
  }
  return b;
}

// One ResNet block for a 16-point tile handled by one wave.
// x_lds: [16][64] f32 input; h_lds: [16][32] f32 scratch (both in LDS);
// w0/w1/ws: f32 weights staged in LDS; b0/b1: global biases.
// out = x@Ws + relu(relu(x)@W0+b0)@W1 + b1  -> out_base[16*32] f32 (global).
__device__ __forceinline__ void resnet_block_wave(
    const float* x_lds, float* h_lds,
    const float* w0, const float* b0, const float* w1, const float* b1,
    const float* ws, float* out_base) {
  const int lane  = threadIdx.x & 31;
  const int nloc  = lane & 15;
  const int mbase = (lane < 16) ? 0 : 8;

  // h1 = relu(x) @ W0 + b0 ; store relu(h1) to LDS (transposed for next A)
  v16h a0r = load_A_lds(x_lds, 64, 0,  true);
  v16h a1r = load_A_lds(x_lds, 64, 32, true);
#pragma unroll
  for (int t = 0; t < 2; ++t) {
    const int n0 = t * 16;
    v8f c = {};
    c = wmma_f16(a0r, load_B_lds(w0, 0,  n0), c);
    c = wmma_f16(a1r, load_B_lds(w0, 32, n0), c);
    const float bias = b0[n0 + nloc];
#pragma unroll
    for (int v = 0; v < 8; ++v) {
      float h = c[v] + bias;
      h = h > 0.f ? h : 0.f;
      h_lds[(mbase + v) * 32 + n0 + nloc] = h;
    }
  }
  __syncthreads();

  // dx = relu(h1) @ W1 + b1 ; out = x @ Ws + dx (accumulated in same C)
  v16h ah = load_A_lds(h_lds, 32, 0,  false);  // already relu'd
  v16h a0 = load_A_lds(x_lds, 64, 0,  false);
  v16h a1 = load_A_lds(x_lds, 64, 32, false);
#pragma unroll
  for (int t = 0; t < 2; ++t) {
    const int n0 = t * 16;
    v8f c = {};
    c = wmma_f16(ah, load_B_lds(w1, 0, n0), c);
    const float bias = b1[n0 + nloc];
#pragma unroll
    for (int v = 0; v < 8; ++v) c[v] += bias;
    c = wmma_f16(a0, load_B_lds(ws, 0,  n0), c);
    c = wmma_f16(a1, load_B_lds(ws, 32, n0), c);
#pragma unroll
    for (int v = 0; v < 8; ++v)
      out_base[(size_t)(mbase + v) * 32 + n0 + nloc] = c[v];
  }
}

// float atomic max via sign-split (init bits must be 0xFF800000 = -inf)
__device__ __forceinline__ void atomicMaxF(float* addr, float v) {
  if (v >= 0.0f) atomicMax((int*)addr, __float_as_int(v));
  else           atomicMin((unsigned int*)addr, __float_as_uint(v));
}

// ---------------------------------------------------------------------------

__global__ void fill_u32_kernel(unsigned int* p, size_t n, unsigned int v) {
  size_t i = (size_t)blockIdx.x * blockDim.x + threadIdx.x;
  const size_t stride = (size_t)gridDim.x * blockDim.x;
  for (; i < n; i += stride) p[i] = v;
}

// stage0: 128 threads = 4 waves, 16 points/wave.
// LDS: per-wave x[16][64]+h[16][32] f32, plus W0|W1|Ws|W_pos|b_pos staged f32.
__global__ void __launch_bounds__(128)
stage0_kernel(const float* __restrict__ p,
              const float* __restrict__ W_pos, const float* __restrict__ b_pos,
              const float* __restrict__ W0, const float* __restrict__ b0,
              const float* __restrict__ W1, const float* __restrict__ b1,
              const float* __restrict__ Ws,
              float* __restrict__ net, int* __restrict__ idx_out) {
  __shared__ __attribute__((aligned(16))) float xh[4 * (16 * 64 + 16 * 32)];
  __shared__ __attribute__((aligned(16))) float wsm[5376];
  const int wave = threadIdx.x >> 5;
  const int lane = threadIdx.x & 31;
  float* x_lds = xh + wave * (16 * 64 + 16 * 32);
  float* h_lds = x_lds + 16 * 64;

  // --- stage weights into LDS: [W0:2048 | W1:1024 | Ws:2048 | Wp:192 | bp:64]
#pragma unroll
  for (int k = 0; k < 10; ++k) {
    const int chunk = threadIdx.x + k * 128;           // 0..1279 (x4 floats)
    const float* src;
    if (chunk < 512)      src = W0 + (size_t)chunk * 4;
    else if (chunk < 768) src = W1 + (size_t)(chunk - 512) * 4;
    else                  src = Ws + (size_t)(chunk - 768) * 4;
    async_copy16(src, wsm + chunk * 4);
  }
  if (threadIdx.x < 64) {
    const int chunk = threadIdx.x;                     // 0..63
    const float* src = (chunk < 48) ? W_pos + (size_t)chunk * 4
                                    : b_pos + (size_t)(chunk - 48) * 4;
    async_copy16(src, wsm + 5120 + chunk * 4);
  }
  async_wait_all();
  __syncthreads();
  const float* wpos = wsm + 5120;   // [3][64]
  const float* bpos = wsm + 5312;   // [64]

  const size_t tile    = (size_t)blockIdx.x * 4 + wave;
  const size_t pt_base = tile * 16;
  const int    row     = lane & 15;
  const size_t pt      = pt_base + row;
  const int    obase   = (lane < 16) ? 0 : 32;

  const float p0 = p[pt * 3 + 0];
  const float p1 = p[pt * 3 + 1];
  const float p2 = p[pt * 3 + 2];
#pragma unroll
  for (int o = 0; o < 32; ++o) {
    const int oo = obase + o;
    x_lds[row * 64 + oo] = bpos[oo] + p0 * wpos[oo] + p1 * wpos[64 + oo]
                                    + p2 * wpos[128 + oo];
  }
  if (lane < 16) {
    // _coords2index: clip(p+1, 0, 1.9999); xi = int(q/2*64) = int(q*32)
    const float q0 = fminf(fmaxf(p0 + 1.f, 0.f), 1.9999f);
    const float q1 = fminf(fmaxf(p1 + 1.f, 0.f), 1.9999f);
    const float q2 = fminf(fmaxf(p2 + 1.f, 0.f), 1.9999f);
    idx_out[pt] = (int)(q0 * 32.f)
                + R_ * ((int)(q1 * 32.f) + R_ * (int)(q2 * 32.f));
  }
  __syncthreads();
  resnet_block_wave(x_lds, h_lds, wsm, b0, wsm + 2048, b1, wsm + 3072,
                    net + pt_base * 32);
}

// scatter-max of net channels into seg[b][vox][32] (seg pre-filled -inf)
__global__ void scatter_kernel(const float* __restrict__ net,
                               const int* __restrict__ idx,
                               float* __restrict__ seg) {
  const size_t t = (size_t)blockIdx.x * blockDim.x + threadIdx.x;
  if (t >= (size_t)B_ * N_ * HID) return;
  const int    ch  = (int)(t & 31);
  const size_t pt  = t >> 5;
  const int    b   = (int)(pt / N_);
  const int    vox = idx[pt];
  atomicMaxF(&seg[(((size_t)b * R3_) + vox) * HID + ch], net[t]);
}

// stage_i: x = concat(net, pooled) DMA'd into LDS, then ResNet block (WMMA).
__global__ void __launch_bounds__(128)
stage_kernel(const float* __restrict__ net_in,
             const float* __restrict__ seg, const int* __restrict__ idx,
             const float* __restrict__ W0, const float* __restrict__ b0,
             const float* __restrict__ W1, const float* __restrict__ b1,
             const float* __restrict__ Ws,
             float* __restrict__ net_out) {
  __shared__ __attribute__((aligned(16))) float xh[4 * (16 * 64 + 16 * 32)];
  __shared__ __attribute__((aligned(16))) float wsm[5120];
  const int wave = threadIdx.x >> 5;
  const int lane = threadIdx.x & 31;
  float* x_lds = xh + wave * (16 * 64 + 16 * 32);
  float* h_lds = x_lds + 16 * 64;

  // --- stage weights into LDS: [W0:2048 | W1:1024 | Ws:2048]
#pragma unroll
  for (int k = 0; k < 10; ++k) {
    const int chunk = threadIdx.x + k * 128;
    const float* src;
    if (chunk < 512)      src = W0 + (size_t)chunk * 4;
    else if (chunk < 768) src = W1 + (size_t)(chunk - 512) * 4;
    else                  src = Ws + (size_t)(chunk - 768) * 4;
    async_copy16(src, wsm + chunk * 4);
  }

  // --- stage activation rows: lanes 0-15 net row -> cols 0..31,
  //                            lanes 16-31 pooled seg row -> cols 32..63
  const size_t tile    = (size_t)blockIdx.x * 4 + wave;
  const size_t pt_base = tile * 16;
  const int    row     = lane & 15;
  const size_t pt      = pt_base + row;

  const float* src;
  if (lane < 16) {
    src = net_in + pt * HID;
  } else {
    const int b   = (int)(pt / N_);
    const int vox = idx[pt];
    src = seg + (((size_t)b * R3_) + vox) * HID;
  }
  float* dst = x_lds + row * 64 + ((lane < 16) ? 0 : 32);
#pragma unroll
  for (int j = 0; j < 8; ++j) async_copy16(src + j * 4, dst + j * 4);

  async_wait_all();
  __syncthreads();
  resnet_block_wave(x_lds, h_lds, wsm, b0, wsm + 2048, b1, wsm + 3072,
                    net_out + pt_base * 32);
}

// final: c = net@W_c + b_c (WMMA), clamp at 0, atomic-umax into
// out[b][feat][vox] (pre-filled 0; nonneg floats order as uints).
__global__ void __launch_bounds__(128)
final_kernel(const float* __restrict__ net, const int* __restrict__ idx,
             const float* __restrict__ W_c, const float* __restrict__ b_c,
             float* __restrict__ out) {
  __shared__ __attribute__((aligned(16))) float xs[4 * 16 * 32];
  __shared__ __attribute__((aligned(16))) float wsm[1024];
  const int wave = threadIdx.x >> 5;
  const int lane = threadIdx.x & 31;
  float* x_lds = xs + wave * (16 * 32);

  // --- stage W_c (1024 floats = 256 chunks, 2 per thread)
#pragma unroll
  for (int k = 0; k < 2; ++k) {
    const int chunk = threadIdx.x + k * 128;
    async_copy16(W_c + (size_t)chunk * 4, wsm + chunk * 4);
  }

  const size_t tile    = (size_t)blockIdx.x * 4 + wave;
  const size_t pt_base = tile * 16;
  const int    row     = lane & 15;
  const size_t pt      = pt_base + row;
  const int    cb      = (lane < 16) ? 0 : 16;

  const float* src = net + pt * HID + cb;
  float*       dst = x_lds + row * 32 + cb;
#pragma unroll
  for (int j = 0; j < 4; ++j) async_copy16(src + j * 4, dst + j * 4);

  async_wait_all();
  __syncthreads();

  v16h a = load_A_lds(x_lds, 32, 0, false);
  const int nloc  = lane & 15;
  const int mbase = (lane < 16) ? 0 : 8;
#pragma unroll
  for (int t = 0; t < 2; ++t) {
    const int n0 = t * 16;
    v8f c = {};
    c = wmma_f16(a, load_B_lds(wsm, 0, n0), c);
    const float bias = b_c[n0 + nloc];
#pragma unroll
    for (int v = 0; v < 8; ++v) {
      const size_t ptm = pt_base + mbase + v;
      const int    b   = (int)(ptm / N_);
      const int    vox = idx[ptm];
      float val = c[v] + bias;
      val = val > 0.f ? val : 0.f;
      unsigned int* dstp = (unsigned int*)&out[(((size_t)b * 32) + (n0 + nloc))
                                               * (size_t)R3_ + vox];
      atomicMax(dstp, __float_as_uint(val));
    }
  }
}

// ---------------------------------------------------------------------------

extern "C" void kernel_launch(void* const* d_in, const int* in_sizes, int n_in,
                              void* d_out, int out_size, void* d_ws, size_t ws_size,
                              hipStream_t stream) {
  const float* p     = (const float*)d_in[0];
  const float* W_pos = (const float*)d_in[1];
  const float* b_pos = (const float*)d_in[2];
  const float* W0    = (const float*)d_in[3];
  const float* b0    = (const float*)d_in[4];
  const float* W1    = (const float*)d_in[5];
  const float* b1    = (const float*)d_in[6];
  const float* Ws    = (const float*)d_in[7];
  const float* W_c   = (const float*)d_in[8];
  const float* b_c   = (const float*)d_in[9];
  float* out = (float*)d_out;

  // workspace: idx[B*N] int, then net[B*N][32] f32 (~53 MB total)
  int*   idx = (int*)d_ws;
  float* net = (float*)((char*)d_ws + (((size_t)B_ * N_ * sizeof(int) + 255ull) & ~255ull));

  // seg scratch [B][R3][32] f32 reuses d_out (identical 134.2 MB footprint)
  float* seg = out;

  const int    stage_blocks = (B_ * N_ / 16) / 4;               // 6250
  const size_t seg_words    = (size_t)B_ * R3_ * HID;           // 33,554,432
  const size_t scat_n       = (size_t)B_ * N_ * HID;            // 12,800,000
  const int    scat_blocks  = (int)((scat_n + 255) / 256);
  const int    fill_blocks  = 4096;

  stage0_kernel<<<stage_blocks, 128, 0, stream>>>(
      p, W_pos, b_pos, W0, b0, W1, b1, Ws, net, idx);

  for (int i = 1; i < 5; ++i) {
    fill_u32_kernel<<<fill_blocks, 256, 0, stream>>>(
        (unsigned int*)seg, seg_words, 0xFF800000u);            // -inf
    scatter_kernel<<<scat_blocks, 256, 0, stream>>>(net, idx, seg);
    stage_kernel<<<stage_blocks, 128, 0, stream>>>(
        net, seg, idx,
        W0 + (size_t)i * 64 * 32, b0 + (size_t)i * 32,
        W1 + (size_t)i * 32 * 32, b1 + (size_t)i * 32,
        Ws + (size_t)i * 64 * 32, net);
  }

  fill_u32_kernel<<<fill_blocks, 256, 0, stream>>>(
      (unsigned int*)out, seg_words, 0u);                       // zeros
  final_kernel<<<stage_blocks, 128, 0, stream>>>(net, idx, W_c, b_c, out);
}
